// RPNHead_25494925869168
// MI455X (gfx1250) — compile-verified, matrix operand order checked by reading, not run on hardware
//
#include <hip/hip_runtime.h>
#include <hip/hip_bf16.h>
#include <stdint.h>

typedef __attribute__((ext_vector_type(16))) __bf16 bf16x16;
typedef __attribute__((ext_vector_type(8)))  float  floatx8;

#define BATCH 4
#define HH    128
#define WW    128
#define CIN   256
#define CMID  512
#define KSTEPS 72        // 2304 / 32
#define NANCH  49152     // 128*128*3
#define MTILE  32        // pixels per workgroup

__device__ __forceinline__ unsigned short f32_to_bf16(float f) {
    unsigned int u = __builtin_bit_cast(unsigned int, f);
    u += 0x7fffu + ((u >> 16) & 1u);           // round-to-nearest-even
    return (unsigned short)(u >> 16);
}

// ---------- prep kernel 1: input f32 -> bf16 (4 elems / thread) ----------
__global__ void convert_input_bf16(const float* __restrict__ in,
                                   unsigned short* __restrict__ out) {
    size_t tid = (size_t)blockIdx.x * blockDim.x + threadIdx.x;
    float4 v = reinterpret_cast<const float4*>(in)[tid];
    ushort4 o;
    o.x = f32_to_bf16(v.x); o.y = f32_to_bf16(v.y);
    o.z = f32_to_bf16(v.z); o.w = f32_to_bf16(v.w);
    reinterpret_cast<ushort4*>(out)[tid] = o;
}

// ---------- prep kernel 2: pack 3x3x256x512 weights into WMMA B layout ----------
// packed[kb][nb][lane][e]; B fragment per lane is one contiguous 32-byte run
__global__ void pack_weights_bf16(const float* __restrict__ w,
                                  unsigned short* __restrict__ wp) {
    unsigned int tid  = blockIdx.x * blockDim.x + threadIdx.x;  // < 72*32*32*16
    unsigned int e    = tid & 15u;
    unsigned int lane = (tid >> 4) & 31u;
    unsigned int nb   = (tid >> 9) & 31u;
    unsigned int kb   = tid >> 14;
    unsigned int tap  = kb >> 3;
    unsigned int c0   = (kb & 7u) << 5;
    unsigned int hi   = lane >> 4;
    unsigned int n    = nb * 16 + (lane & 15u);
    // ISA 16-bit B layout: lanes 0-15 hold K 0..7 & 16..23, lanes 16-31 hold K 8..15 & 24..31
    unsigned int klocal = (e < 8) ? (hi * 8 + e) : (16 + hi * 8 + (e - 8));
    float v = w[((size_t)(tap * 256 + c0 + klocal)) * 512 + n];
    wp[tid] = f32_to_bf16(v);
}

// ---------- fused conv3x3(relu) + 1x1 heads + softmax, M-tile = 32 pixels ----------
__global__ __launch_bounds__(256) void rpn_fused_kernel(
    const unsigned short* __restrict__ in_bf,   // [B,H,W,256] bf16
    const unsigned short* __restrict__ w_pack,  // [72][32][32][16] bf16
    const float* __restrict__ b_shared,         // [512]
    const float* __restrict__ w_cls,            // [512,6]
    const float* __restrict__ b_cls,            // [6]
    const float* __restrict__ w_delta,          // [512,12]
    const float* __restrict__ b_delta,          // [12]
    float* __restrict__ out)                    // logits | probs | deltas
{
    // Overlapped LDS: phase 1 = bf16 halo slab (3x34x256 = 52.2 KB),
    //                 phase 2 = f32 activation tile (32x512 = 64 KB)
    __shared__ __align__(16) unsigned char smem_raw[65536];
    unsigned short* slab = reinterpret_cast<unsigned short*>(smem_raw);
    float*          shm  = reinterpret_cast<float*>(smem_raw);

    const int t    = threadIdx.x;
    const int wave = t >> 5;
    const int lane = t & 31;
    const int half = lane >> 4;
    const int l16  = lane & 15;

    const int tile = blockIdx.x;          // 2048 tiles = B * H * (W/32)
    const int w0   = (tile & 3) << 5;
    const int h    = (tile >> 2) & 127;
    const int b    = tile >> 9;

    // ---- stage halo slab (rows h-1..h+1, cols w0-1..w0+32, 256 ch), zero pad ----
    for (int q = t; q < 102 * 32; q += 256) {     // 16-byte chunks
        int pos = q >> 5;                          // 0..101 (3*34)
        int cq  = q & 31;
        int dyi = pos / 34;
        int wxi = pos - dyi * 34;
        int hy  = h + dyi - 1;
        int wx  = w0 + wxi - 1;
        float4 val = make_float4(0.f, 0.f, 0.f, 0.f);
        if (hy >= 0 && hy < HH && wx >= 0 && wx < WW) {
            size_t gi = ((size_t)(b * HH + hy) * WW + wx) * CIN + (size_t)cq * 8;
            val = *reinterpret_cast<const float4*>(in_bf + gi);
        }
        *reinterpret_cast<float4*>(&slab[pos * 256 + cq * 8]) = val;
    }
    __syncthreads();

    union Frag { bf16x16 v; float4 f4[2]; };
    floatx8 acc0[4] = {};   // pixels 0..15 of tile
    floatx8 acc1[4] = {};   // pixels 16..31 of tile

    // ---- implicit-GEMM K loop: 72 steps x (8 x v_wmma_f32_16x16x32_bf16) ----
    // Each B fragment is reused for both M halves -> B L2 traffic halved vs MTILE=16.
    for (int kb = 0; kb < KSTEPS; ++kb) {
        int tap = kb >> 3;
        int dy  = tap / 3;
        int dx  = tap - dy * 3;
        int c0  = (kb & 7) << 5;

        // A fragments (16x32 bf16): lane<16 -> K 0..7 & 16..23, lane>=16 -> K 8..15 & 24..31
        Frag af0, af1;
        int base0 = (dy * 34 + dx + l16) * 256 + c0 + half * 8;
        af0.f4[0] = *reinterpret_cast<const float4*>(&slab[base0]);
        af0.f4[1] = *reinterpret_cast<const float4*>(&slab[base0 + 16]);
        int base1 = base0 + 16 * 256;            // pixels +16
        af1.f4[0] = *reinterpret_cast<const float4*>(&slab[base1]);
        af1.f4[1] = *reinterpret_cast<const float4*>(&slab[base1 + 16]);

        const unsigned short* wbase =
            w_pack + ((size_t)(kb * 32 + wave * 4) * 32 + lane) * 16;
#pragma unroll
        for (int j = 0; j < 4; ++j) {
            Frag bfr;
            const float4* wp4 = reinterpret_cast<const float4*>(wbase + (size_t)j * 512);
            bfr.f4[0] = wp4[0];
            bfr.f4[1] = wp4[1];
            acc0[j] = __builtin_amdgcn_wmma_f32_16x16x32_bf16(
                false, af0.v, false, bfr.v, (short)0, acc0[j], false, false);
            acc1[j] = __builtin_amdgcn_wmma_f32_16x16x32_bf16(
                false, af1.v, false, bfr.v, (short)0, acc1[j], false, false);
        }
    }
    __syncthreads();   // all slab reads done before shm overwrites the same LDS

    // ---- bias + ReLU, scatter 32x512 tile to LDS (C layout: M = r + 8*half) ----
#pragma unroll
    for (int j = 0; j < 4; ++j) {
        int n = wave * 64 + j * 16 + l16;
        float bias = b_shared[n];
#pragma unroll
        for (int r = 0; r < 8; ++r) {
            float v0 = acc0[j][r] + bias;
            float v1 = acc1[j][r] + bias;
            shm[(r + 8 * half) * 512 + n]        = fmaxf(v0, 0.f);
            shm[(16 + r + 8 * half) * 512 + n]   = fmaxf(v1, 0.f);
        }
    }
    __syncthreads();

    // ---- fused 1x1 heads + softmax: each wave handles 4 pixels ----
    const size_t PROB_OFF  = (size_t)BATCH * NANCH * 2;  // 393216
    const size_t DELTA_OFF = 2 * PROB_OFF;               // 786432
    for (int pi = 0; pi < 4; ++pi) {
        int p = wave * 4 + pi;
        float s18[18];
#pragma unroll
        for (int o = 0; o < 18; ++o) s18[o] = 0.f;
        for (int k = lane; k < 512; k += 32) {
            float s = shm[p * 512 + k];
#pragma unroll
            for (int o = 0; o < 6; ++o)  s18[o]     += s * w_cls[k * 6 + o];
#pragma unroll
            for (int o = 0; o < 12; ++o) s18[6 + o] += s * w_delta[k * 12 + o];
        }
#pragma unroll
        for (int o = 0; o < 18; ++o) {
#pragma unroll
            for (int off = 16; off > 0; off >>= 1)
                s18[o] += __shfl_xor(s18[o], off, 32);
        }
        if (lane < 3) {  // lane = anchor index
            int wpix = w0 + p;
            int anchor = (h * WW + wpix) * 3 + lane;
            size_t idx2 = ((size_t)b * NANCH + anchor) * 2;
            size_t idx4 = ((size_t)b * NANCH + anchor) * 4;
            float l0 = s18[2 * lane]     + b_cls[2 * lane];
            float l1 = s18[2 * lane + 1] + b_cls[2 * lane + 1];
            out[idx2]     = l0;
            out[idx2 + 1] = l1;
            float mx = fmaxf(l0, l1);
            float e0 = __expf(l0 - mx);
            float e1 = __expf(l1 - mx);
            float inv = 1.f / (e0 + e1);
            out[PROB_OFF + idx2]     = e0 * inv;
            out[PROB_OFF + idx2 + 1] = e1 * inv;
#pragma unroll
            for (int jj = 0; jj < 4; ++jj)
                out[DELTA_OFF + idx4 + jj] = s18[6 + 4 * lane + jj] + b_delta[4 * lane + jj];
        }
    }
}

extern "C" void kernel_launch(void* const* d_in, const int* in_sizes, int n_in,
                              void* d_out, int out_size, void* d_ws, size_t ws_size,
                              hipStream_t stream) {
    (void)in_sizes; (void)n_in; (void)out_size; (void)ws_size;
    const float* inputs   = (const float*)d_in[0];
    const float* w_shared = (const float*)d_in[1];
    const float* b_sh     = (const float*)d_in[2];
    const float* w_cls    = (const float*)d_in[3];
    const float* b_cls    = (const float*)d_in[4];
    const float* w_delta  = (const float*)d_in[5];
    const float* b_delta  = (const float*)d_in[6];
    float* out = (float*)d_out;

    unsigned short* in_bf  = (unsigned short*)d_ws;                       // 32 MB
    unsigned short* w_pack = (unsigned short*)((char*)d_ws +
                              (size_t)BATCH * HH * WW * CIN * 2);         // +2.36 MB

    convert_input_bf16<<<16384, 256, 0, stream>>>(inputs, in_bf);    // 16.7M elems / 4
    pack_weights_bf16 <<<4608,  256, 0, stream>>>(w_shared, w_pack); // 1.18M packed elems
    rpn_fused_kernel  <<<2048,  256, 0, stream>>>(in_bf, w_pack, b_sh,
                                                  w_cls, b_cls, w_delta, b_delta, out);
}
